// MLAAttention_77137612636297
// MI455X (gfx1250) — matrix operand order repaired and checked by reading, not compiled, not run on hardware
//
#include <hip/hip_runtime.h>
#include <hip/hip_bf16.h>
#include <stdint.h>

#define NH     16
#define NOPE   128
#define ROPED  64
#define VDIM   128
#define LORA   512
#define DMODEL 2048
#define TTOK   256
#define SKV    32768
#define KSEL   2048
#define CDIM   576      // LORA + ROPED  (combined latent+rope key dim)
#define QHD    192      // NOPE + ROPED
#define TB     4        // tokens per block in projection kernels

static constexpr float SCALE = 0.07216878364870322f;   // 1/sqrt(192)

// ---------- types for WMMA ----------
typedef __attribute__((ext_vector_type(16))) __bf16 v16bf;
typedef __attribute__((ext_vector_type(8)))  float  v8f;

static __device__ inline v16bf mkbf(uint4 lo, uint4 hi) {
    union { uint4 q[2]; v16bf v; } u;
    u.q[0] = lo; u.q[1] = hi;
    return u.v;
}

static __device__ inline v8f wmma_bf16(v16bf a, v16bf b, v8f c) {
    // D = A(16x32 bf16) * B(32x16 bf16) + C(16x16 f32)
    return __builtin_amdgcn_wmma_f32_16x16x32_bf16(
        /*neg_a=*/false, a, /*neg_b=*/false, b,
        /*c_mod=*/(short)0, c, /*reuse_a=*/false, /*reuse_b=*/false);
}

// fp32 -> bf16 round-to-nearest-even
static __device__ inline uint16_t f2bf(float f) {
    uint32_t u = __float_as_uint(f);
    u += 0x7FFFu + ((u >> 16) & 1u);
    return (uint16_t)(u >> 16);
}

// ================= K0: combined bf16 key cache: [512 latent | 64 roped pe] =================
__global__ __launch_bounds__(576)
void k_prep_kc(const float* __restrict__ kv_c, const float* __restrict__ k_pe,
               uint16_t* __restrict__ kc) {
    int s = blockIdx.x, tid = threadIdx.x;
    uint16_t* row = kc + (size_t)s * CDIM;
    if (tid < LORA) {
        row[tid] = f2bf(kv_c[(size_t)s * LORA + tid]);
    } else {
        int j  = tid - LORA;
        int jj = j & 31;
        const float* pe = k_pe + (size_t)s * ROPED;
        float inv = __powf(10000.0f, -(float)jj / 32.0f);
        float f   = (float)s * inv;
        float sn, cs; __sincosf(f, &sn, &cs);
        float x1 = pe[jj], x2 = pe[jj + 32];
        row[LORA + j] = f2bf((j < 32) ? (x1 * cs - x2 * sn) : (x2 * cs + x1 * sn));
    }
}

// ================= K1a: q = x @ Wq  -> qfull[T][H*192] f32 (4 tokens/block) =================
__global__ __launch_bounds__(256)
void k_qproj(const float* __restrict__ x, const float* __restrict__ Wq,
             float* __restrict__ qf) {
    __shared__ float xs[TB][DMODEL];               // 32 KB
    int t0  = blockIdx.x * TB;
    int col = blockIdx.y * 256 + threadIdx.x;      // 0..3071
    for (int i = threadIdx.x; i < TB * DMODEL; i += 256)
        xs[i >> 11][i & (DMODEL - 1)] = x[(size_t)(t0 + (i >> 11)) * DMODEL + (i & (DMODEL - 1))];
    __syncthreads();
    float acc[TB] = {0.f, 0.f, 0.f, 0.f};
    for (int i = 0; i < DMODEL; ++i) {
        float w = Wq[(size_t)i * (NH * QHD) + col];
        #pragma unroll
        for (int tt = 0; tt < TB; ++tt) acc[tt] += xs[tt][i] * w;
    }
    #pragma unroll
    for (int tt = 0; tt < TB; ++tt)
        qf[(size_t)(t0 + tt) * (NH * QHD) + col] = acc[tt];
}

// ================= K1b: absorb W_UK + rope q_pe -> qc[T][H][576] bf16 (4 tokens/block) =====
__global__ __launch_bounds__(576)
void k_qcomb(const float* __restrict__ qf, const float* __restrict__ wuk,
             const int* __restrict__ pos_q, uint16_t* __restrict__ qc) {
    int t0 = blockIdx.x * TB, h = blockIdx.y, tid = threadIdx.x;
    __shared__ float qn[TB][QHD];
    for (int i = tid; i < TB * QHD; i += 576) {
        int tt = i / QHD, e = i % QHD;
        qn[tt][e] = qf[((size_t)(t0 + tt) * NH + h) * QHD + e];
    }
    __syncthreads();
    if (tid < LORA) {
        const float* w = wuk + (size_t)h * NOPE * LORA + tid;
        float acc[TB] = {0.f, 0.f, 0.f, 0.f};
        for (int n = 0; n < NOPE; ++n) {
            float wv = w[(size_t)n * LORA];
            #pragma unroll
            for (int tt = 0; tt < TB; ++tt) acc[tt] += qn[tt][n] * wv;
        }
        #pragma unroll
        for (int tt = 0; tt < TB; ++tt)
            qc[((size_t)(t0 + tt) * NH + h) * CDIM + tid] = f2bf(acc[tt]);
    } else {
        int j = tid - LORA, jj = j & 31;
        float inv = __powf(10000.0f, -(float)jj / 32.0f);
        #pragma unroll
        for (int tt = 0; tt < TB; ++tt) {
            float f = (float)pos_q[t0 + tt] * inv;
            float sn, cs; __sincosf(f, &sn, &cs);
            float x1 = qn[tt][NOPE + jj], x2 = qn[tt][NOPE + jj + 32];
            qc[((size_t)(t0 + tt) * NH + h) * CDIM + LORA + j] =
                f2bf((j < 32) ? (x1 * cs - x2 * sn) : (x2 * cs + x1 * sn));
        }
    }
}

// ================= K2: scores[T][16][2048] = Q(16x576) . Kgathered(576x2048) =================
// grid (T, 16); 8 waves x 16 keys = 128 keys per block. Each wave: one 16x16 WMMA tile.
__global__ __launch_bounds__(256)
void k_scores(const uint16_t* __restrict__ qc, const uint16_t* __restrict__ kc,
              const int* __restrict__ topk, float* __restrict__ sc) {
    __shared__ uint16_t qtile[16 * CDIM];          // 18 KB
    int t = blockIdx.x, kgrp = blockIdx.y, tid = threadIdx.x;

    const uint4* qsrc = (const uint4*)(qc + (size_t)t * 16 * CDIM);
    uint4* qdst = (uint4*)qtile;
    for (int i = tid; i < 16 * CDIM / 8; i += 256) qdst[i] = qsrc[i];
    __syncthreads();

    int wave = tid >> 5, lane = tid & 31;
    int half = lane >> 4, ln = lane & 15;
    int key  = (kgrp * 8 + wave) * 16 + ln;        // B column n = lane%16
    int idx  = topk[(size_t)t * KSEL + key];
    const uint4* kr = (const uint4*)(kc + (size_t)idx * CDIM);
    const uint16_t* qrow = qtile + ln * CDIM;      // A row m = lane%16 (head)

    v8f acc = {0.f, 0.f, 0.f, 0.f, 0.f, 0.f, 0.f, 0.f};
    #pragma unroll
    for (int s = 0; s < CDIM / 32; ++s) {          // 18 WMMA steps over K=576
        int fb = s * 32;
        // A 16x32: lanes<16 hold K 0-7|16-23, lanes>=16 hold K 8-15|24-31
        uint4 a0 = *(const uint4*)(qrow + fb + 8 * half);
        uint4 a1 = *(const uint4*)(qrow + fb + 16 + 8 * half);
        // B 32x16: lanes<16 hold K 0-7|16-23, lanes>=16 hold K 8-15|24-31
        uint4 b0 = kr[(fb >> 3) + half];
        uint4 b1 = kr[(fb >> 3) + 2 + half];
        acc = wmma_bf16(mkbf(a0, a1), mkbf(b0, b1), acc);
    }
    // C: VGPR r -> head = r + 8*half, N = lane%16
    float* out = sc + ((size_t)t * 16 + 8 * half) * KSEL + key;
    #pragma unroll
    for (int r = 0; r < 8; ++r)
        out[(size_t)r * KSEL] = acc[r] * SCALE;
}

// ================= K3: softmax over K=2048, emit bf16 probs =================
__global__ __launch_bounds__(256)
void k_softmax(const float* __restrict__ sc, uint16_t* __restrict__ probs) {
    __shared__ float red[256];
    int row = blockIdx.x, tid = threadIdx.x;       // row = t*16 + h
    const float* s = sc + (size_t)row * KSEL;
    float v[8], mx = -3.4e38f;
    #pragma unroll
    for (int i = 0; i < 8; ++i) { v[i] = s[tid + i * 256]; mx = fmaxf(mx, v[i]); }
    red[tid] = mx; __syncthreads();
    for (int o = 128; o > 0; o >>= 1) { if (tid < o) red[tid] = fmaxf(red[tid], red[tid + o]); __syncthreads(); }
    mx = red[0]; __syncthreads();
    float sum = 0.f;
    #pragma unroll
    for (int i = 0; i < 8; ++i) { v[i] = __expf(v[i] - mx); sum += v[i]; }
    red[tid] = sum; __syncthreads();
    for (int o = 128; o > 0; o >>= 1) { if (tid < o) red[tid] += red[tid + o]; __syncthreads(); }
    float inv = 1.0f / red[0];
    uint16_t* p = probs + (size_t)row * KSEL;
    #pragma unroll
    for (int i = 0; i < 8; ++i) p[tid + i * 256] = f2bf(v[i] * inv);
}

// ================= K4: o_lat[T][16][512] = P(16x2048) . KVsel(2048x512) =================
// grid (T); chunks of 32 keys staged TRANSPOSED in LDS (b64 stores: 4 keys packed)
// so the key dim becomes the per-lane contraction dim of the B operand.
// 8 waves x 4 feat tiles = 512 feats.
__global__ __launch_bounds__(256)
void k_olat(const uint16_t* __restrict__ probs, const uint16_t* __restrict__ kc,
            const int* __restrict__ topk, float* __restrict__ olat) {
    __shared__ uint16_t kvT[LORA * 32];            // [feat][key] 32 KB
    int t = blockIdx.x, tid = threadIdx.x;
    int wave = tid >> 5, lane = tid & 31, half = lane >> 4, ln = lane & 15;
    int fseg = tid >> 3;                           // 0..31 -> feats f0 = fseg*16
    int kgrp = tid & 7;                            // 0..7  -> keys  kk0 = kgrp*4
    int f0 = fseg * 16, kk0 = kgrp * 4;
    const uint16_t* prow = probs + ((size_t)t * 16 + ln) * KSEL;   // A row = head

    v8f acc[4];
    #pragma unroll
    for (int i = 0; i < 4; ++i) acc[i] = (v8f){0.f,0.f,0.f,0.f,0.f,0.f,0.f,0.f};

    for (int kb = 0; kb < KSEL; kb += 32) {
        // ---- prefetch next chunk's gathered rows (global_prefetch_b8) ----
        if (kb + 32 < KSEL) {
            #pragma unroll
            for (int r = 0; r < 4; ++r) {
                int pidx = topk[(size_t)t * KSEL + kb + 32 + kk0 + r];
                __builtin_prefetch(kc + (size_t)pidx * CDIM + f0, 0, 1);
            }
        }
        // ---- stage 4 keys x 16 feats per thread, transposed, packed b64 stores ----
        uint32_t gw[4][8];
        #pragma unroll
        for (int r = 0; r < 4; ++r) {
            int idx = topk[(size_t)t * KSEL + kb + kk0 + r];
            const uint4* src = (const uint4*)(kc + (size_t)idx * CDIM) + fseg * 2;
            uint4 lo = src[0], hi = src[1];
            gw[r][0] = lo.x; gw[r][1] = lo.y; gw[r][2] = lo.z; gw[r][3] = lo.w;
            gw[r][4] = hi.x; gw[r][5] = hi.y; gw[r][6] = hi.z; gw[r][7] = hi.w;
        }
        #pragma unroll
        for (int f = 0; f < 16; ++f) {
            int dw = f >> 1, sh = (f & 1) * 16;
            uint32_t v0 = (gw[0][dw] >> sh) & 0xFFFFu;
            uint32_t v1 = (gw[1][dw] >> sh) & 0xFFFFu;
            uint32_t v2 = (gw[2][dw] >> sh) & 0xFFFFu;
            uint32_t v3 = (gw[3][dw] >> sh) & 0xFFFFu;
            uint2 pk; pk.x = v0 | (v1 << 16); pk.y = v2 | (v3 << 16);
            *(uint2*)(kvT + (f0 + f) * 32 + kk0) = pk;   // ds_store_b64
        }
        __syncthreads();
        // ---- A: probs tile 16 heads x 32 keys ----
        uint4 a0 = *(const uint4*)(prow + kb + 8 * half);
        uint4 a1 = *(const uint4*)(prow + kb + 16 + 8 * half);
        v16bf A = mkbf(a0, a1);
        #pragma unroll
        for (int tI = 0; tI < 4; ++tI) {
            int ft = (wave * 4 + tI) * 16;
            // B: 32 keys x 16 feats; lane column n = feat, keys contiguous in kvT row
            const uint16_t* brow = kvT + (ft + ln) * 32 + 8 * half;
            uint4 b0 = *(const uint4*)brow;
            uint4 b1 = *(const uint4*)(brow + 16);
            acc[tI] = wmma_bf16(A, mkbf(b0, b1), acc[tI]);
        }
        __syncthreads();
    }
    #pragma unroll
    for (int tI = 0; tI < 4; ++tI) {
        int ft = (wave * 4 + tI) * 16;
        float* o = olat + ((size_t)t * 16 + 8 * half) * LORA + ft + ln;
        #pragma unroll
        for (int r = 0; r < 8; ++r)
            o[(size_t)r * LORA] = acc[tI][r];
    }
}

// ================= K5a: ov = einsum(o_lat, W_UV) -> [T][2048] (4 tokens/block) =================
__global__ __launch_bounds__(256)
void k_ovproj(const float* __restrict__ olat, const float* __restrict__ wuv,
              float* __restrict__ ov) {
    __shared__ float olds[TB][2][LORA];            // 16 KB (block covers 2 heads)
    int t0 = blockIdx.x * TB;
    int hbase = blockIdx.y * 2;                    // 256 cols = 2 heads of 128
    for (int i = threadIdx.x; i < TB * 2 * LORA; i += 256) {
        int tt = i >> 10, hh = (i >> 9) & 1, l = i & (LORA - 1);
        olds[tt][hh][l] = olat[((size_t)(t0 + tt) * NH + hbase + hh) * LORA + l];
    }
    __syncthreads();
    int col = blockIdx.y * 256 + threadIdx.x;      // h*128 + v
    int hh = (threadIdx.x >> 7) & 1;
    const float* w = wuv + (size_t)col * LORA;
    float acc[TB] = {0.f, 0.f, 0.f, 0.f};
    for (int l = 0; l < LORA; ++l) {
        float wv = w[l];
        #pragma unroll
        for (int tt = 0; tt < TB; ++tt) acc[tt] += olds[tt][hh][l] * wv;
    }
    #pragma unroll
    for (int tt = 0; tt < TB; ++tt)
        ov[(size_t)(t0 + tt) * DMODEL + col] = acc[tt];
}

// ================= K5b: out = ov @ Wo (4 tokens/block) =================
__global__ __launch_bounds__(256)
void k_outproj(const float* __restrict__ ov, const float* __restrict__ Wo,
               float* __restrict__ out) {
    __shared__ float vs[TB][DMODEL];               // 32 KB
    int t0 = blockIdx.x * TB;
    int d = blockIdx.y * 256 + threadIdx.x;
    for (int i = threadIdx.x; i < TB * DMODEL; i += 256)
        vs[i >> 11][i & (DMODEL - 1)] = ov[(size_t)(t0 + (i >> 11)) * DMODEL + (i & (DMODEL - 1))];
    __syncthreads();
    float acc[TB] = {0.f, 0.f, 0.f, 0.f};
    for (int c = 0; c < DMODEL; ++c) {
        float w = Wo[(size_t)c * DMODEL + d];
        #pragma unroll
        for (int tt = 0; tt < TB; ++tt) acc[tt] += vs[tt][c] * w;
    }
    #pragma unroll
    for (int tt = 0; tt < TB; ++tt)
        out[(size_t)(t0 + tt) * DMODEL + d] = acc[tt];
}

// ================= workspace layout (bytes) =================
static constexpr size_t OFF_KC     = 0;                                    // S*576 bf16
static constexpr size_t OFF_QFULL  = OFF_KC     + (size_t)SKV * CDIM * 2;  // T*3072 f32
static constexpr size_t OFF_QC     = OFF_QFULL  + (size_t)TTOK * NH * QHD * 4;
static constexpr size_t OFF_SCORES = OFF_QC     + (size_t)TTOK * NH * CDIM * 2;
static constexpr size_t OFF_PROBS  = OFF_SCORES + (size_t)TTOK * NH * KSEL * 4;
static constexpr size_t OFF_OLAT   = OFF_PROBS  + (size_t)TTOK * NH * KSEL * 2;
static constexpr size_t OFF_OV     = OFF_OLAT   + (size_t)TTOK * NH * LORA * 4;

extern "C" void kernel_launch(void* const* d_in, const int* in_sizes, int n_in,
                              void* d_out, int out_size, void* d_ws, size_t ws_size,
                              hipStream_t stream) {
    const float* x     = (const float*)d_in[0];
    const float* Wq    = (const float*)d_in[1];
    const float* W_UK  = (const float*)d_in[2];
    const float* W_UV  = (const float*)d_in[3];
    const float* Wo    = (const float*)d_in[4];
    const float* kv_c  = (const float*)d_in[5];
    const float* k_pe  = (const float*)d_in[6];
    const int*   topk  = (const int*)d_in[7];
    const int*   pos_q = (const int*)d_in[8];
    float* out = (float*)d_out;

    char* ws = (char*)d_ws;
    uint16_t* kc     = (uint16_t*)(ws + OFF_KC);
    float*    qfull  = (float*)   (ws + OFF_QFULL);
    uint16_t* qc     = (uint16_t*)(ws + OFF_QC);
    float*    scores = (float*)   (ws + OFF_SCORES);
    uint16_t* probs  = (uint16_t*)(ws + OFF_PROBS);
    float*    olat   = (float*)   (ws + OFF_OLAT);
    float*    ov     = (float*)   (ws + OFF_OV);

    k_prep_kc <<<SKV, 576, 0, stream>>>(kv_c, k_pe, kc);
    k_qproj   <<<dim3(TTOK / TB, (NH * QHD) / 256), 256, 0, stream>>>(x, Wq, qfull);
    k_qcomb   <<<dim3(TTOK / TB, NH), 576, 0, stream>>>(qfull, W_UK, pos_q, qc);
    k_scores  <<<dim3(TTOK, KSEL / 128), 256, 0, stream>>>(qc, kc, topk, scores);
    k_softmax <<<TTOK * NH, 256, 0, stream>>>(scores, probs);
    k_olat    <<<TTOK, 256, 0, stream>>>(probs, kc, topk, olat);
    k_ovproj  <<<dim3(TTOK / TB, DMODEL / 256), 256, 0, stream>>>(olat, W_UV, ov);
    k_outproj <<<dim3(TTOK / TB, DMODEL / 256), 256, 0, stream>>>(ov, Wo, out);
}
